// EntityConsolidationHead_27960237097197
// MI455X (gfx1250) — compile-verified, matrix operand order checked by reading, not run on hardware
//
#include <hip/hip_runtime.h>
#include <hip/hip_bf16.h>

// ---------------------------------------------------------------------------
// EntityConsolidationHead for MI455X (gfx1250, wave32, WMMA bf16)
//
// logits[p] = W2 . gelu( [a,b,|a-b|,a*b] @ W1 + b1 ) + b2
// loss      = mean BCE-with-logits
//
// Strategy: bf16 WMMA (v_wmma_f32_16x16x32_bf16) with f32 accumulation.
// W1 is pre-transposed+converted to bf16 once into d_ws so LDS fragment
// loads are contiguous 16B runs matching the CDNA5 16-bit A/B layouts.
// B-chunk staging uses gfx1250 async global->LDS copies (ASYNCcnt) when the
// builtins are available, avoiding the VGPR round-trip.
// ---------------------------------------------------------------------------

typedef __attribute__((ext_vector_type(16))) __bf16 v16bf;
typedef __attribute__((ext_vector_type(8)))  float  v8f;

union Frag {
    v16bf v;
    uint4 q[2];
};

#define H       768
#define K4      3072            // 4*H
#define BM      32              // pair rows per block
#define NCHUNK  96              // K4 / 32
#define NT_PER_WAVE 6           // 48 N-tiles / 8 waves

// LDS layout (single 64,256B arena, <= 64KB):
//   [0,      2560)  A chunk  : 32 rows x stride 40 ushort (bf16)
//   [2560,   2688)  iaS      : 32 int
//   [2688,   2816)  ibS      : 32 int
//   [2816,  64256)  B chunk  : 768 rows x stride 40 ushort (bf16)
//   [2816,  19200)  red      : 32 x 128 float   (overlays B after GEMM)
#define SMEM_BYTES 64256
#define A_OFF   0
#define IA_OFF  2560
#define IB_OFF  2688
#define B_OFF   2816
#define LSTRIDE 40              // padded bf16 row stride (80B) for bank spread

// ---- gfx1250 async direct-to-LDS path (guarded; falls back to sync fill) --
#if defined(__gfx1250__) && __has_builtin(__builtin_amdgcn_global_load_async_to_lds_b128)
#define USE_ASYNC_LDS 1
typedef int v4i_t __attribute__((vector_size(16)));          // gcc-vector int4
typedef __attribute__((address_space(1))) void  GVoid;       // __device__ void
typedef __attribute__((address_space(3))) void  LVoid;       // __shared__ void
typedef __attribute__((address_space(1))) v4i_t GV4i;        // __device__ int4
typedef __attribute__((address_space(3))) v4i_t LV4i;        // __shared__ int4
#if __has_builtin(__builtin_amdgcn_s_wait_asynccnt)
#define WAIT_ASYNC0() __builtin_amdgcn_s_wait_asynccnt(0)
#else
#define WAIT_ASYNC0() asm volatile("s_wait_asynccnt 0" ::: "memory")
#endif
#else
#define USE_ASYNC_LDS 0
#endif

__device__ __forceinline__ unsigned f32_to_bf16(float f) {
    unsigned u = __float_as_uint(f);
    u += 0x7FFFu + ((u >> 16) & 1u);      // round-to-nearest-even
    return u >> 16;
}

// ---------------------------------------------------------------------------
// Kernel 0: W1 [3072,768] f32  ->  W1t [768,3072] bf16 (transposed) in d_ws
// ---------------------------------------------------------------------------
__global__ void convert_w1_kernel(const float* __restrict__ W1,
                                  unsigned short* __restrict__ W1t) {
    int idx = blockIdx.x * blockDim.x + threadIdx.x;
    if (idx < K4 * H) {
        int k = idx / H;
        int n = idx - k * H;
        W1t[(size_t)n * K4 + k] = (unsigned short)f32_to_bf16(W1[idx]);
    }
}

// ---------------------------------------------------------------------------
// Kernel 1: fused gather + feature build + WMMA GEMM + GELU + W2 dot
// grid = ceil(P/32) blocks of 256 threads (8 wave32)
// ---------------------------------------------------------------------------
__global__ void __launch_bounds__(256)
pair_head_gemm_kernel(const float* __restrict__ emb,
                      const int*   __restrict__ pa,
                      const int*   __restrict__ pb,
                      const unsigned short* __restrict__ W1t,
                      const float* __restrict__ b1,
                      const float* __restrict__ W2,
                      const float* __restrict__ b2,
                      float* __restrict__ logits,
                      int P) {
    __shared__ __align__(16) unsigned char smem[SMEM_BYTES];
    unsigned short* Alds = (unsigned short*)(smem + A_OFF);
    int*            iaS  = (int*)(smem + IA_OFF);
    int*            ibS  = (int*)(smem + IB_OFF);
    unsigned short* Blds = (unsigned short*)(smem + B_OFF);
    float*          red  = (float*)(smem + B_OFF);

    const int tid  = threadIdx.x;
    const int wv   = tid >> 5;          // wave 0..7
    const int lane = tid & 31;
    const int lm   = lane & 15;
    const bool hi  = lane >= 16;
    const int kb   = hi ? 8 : 0;        // bf16 K-run base per CDNA5 layout
    const int r0   = blockIdx.x * BM;

    if (tid < BM) {
        int r = r0 + tid;
        if (r >= P) r = P - 1;          // clamp tail (writes guarded later)
        iaS[tid] = pa[r];
        ibS[tid] = pb[r];
    }
    __syncthreads();

    v8f acc[2][NT_PER_WAVE];
    #pragma unroll
    for (int mt = 0; mt < 2; ++mt)
        #pragma unroll
        for (int nt = 0; nt < NT_PER_WAVE; ++nt)
            acc[mt][nt] = (v8f){0.f, 0.f, 0.f, 0.f, 0.f, 0.f, 0.f, 0.f};

    for (int c = 0; c < NCHUNK; ++c) {
        const int seg  = c / 24;            // which feature segment
        const int col0 = (c % 24) * 32;     // column within H
        const int kc   = c * 32;            // K offset into W1t rows

        // ---- fill B chunk: W1t[n, kc..kc+31] -> Blds[n][0..31] ------------
        // gfx1250 path: async global->LDS B128 copies (no VGPR staging),
        // overlapping with the VALU feature construction for the A chunk.
#if USE_ASYNC_LDS
        #pragma unroll 4
        for (int i = tid; i < H * 4; i += 256) {
            const int n  = i >> 2;
            const int kp = i & 3;
            GV4i* g = (GV4i*)(GVoid*)(W1t + (size_t)n * K4 + kc + kp * 8);
            LV4i* l = (LV4i*)(LVoid*)(Blds + n * LSTRIDE + kp * 8);
            __builtin_amdgcn_global_load_async_to_lds_b128(g, l, 0, 0);
        }
#else
        for (int i = tid; i < H * 4; i += 256) {
            const int n  = i >> 2;
            const int kp = i & 3;
            uint4 v = *(const uint4*)(W1t + (size_t)n * K4 + kc + kp * 8);
            *(uint4*)(Blds + n * LSTRIDE + kp * 8) = v;
        }
#endif

        // ---- fill A chunk: feats[r0..r0+31, col0..col0+31] as bf16 --------
        {
            const int row  = tid >> 3;
            const int colq = (tid & 7) << 2;
            const int col  = col0 + colq;
            float4 av = {0.f, 0.f, 0.f, 0.f};
            float4 bv = {0.f, 0.f, 0.f, 0.f};
            if (seg != 1) av = *(const float4*)(emb + (size_t)iaS[row] * H + col);
            if (seg != 0) bv = *(const float4*)(emb + (size_t)ibS[row] * H + col);
            float f0, f1, f2, f3;
            if (seg == 0)      { f0 = av.x; f1 = av.y; f2 = av.z; f3 = av.w; }
            else if (seg == 1) { f0 = bv.x; f1 = bv.y; f2 = bv.z; f3 = bv.w; }
            else if (seg == 2) { f0 = fabsf(av.x - bv.x); f1 = fabsf(av.y - bv.y);
                                 f2 = fabsf(av.z - bv.z); f3 = fabsf(av.w - bv.w); }
            else               { f0 = av.x * bv.x; f1 = av.y * bv.y;
                                 f2 = av.z * bv.z; f3 = av.w * bv.w; }
            uint2 pk;
            pk.x = f32_to_bf16(f0) | (f32_to_bf16(f1) << 16);
            pk.y = f32_to_bf16(f2) | (f32_to_bf16(f3) << 16);
            *(uint2*)(Alds + row * LSTRIDE + colq) = pk;
        }

        // prefetch next chunk of W1t into cache (global_prefetch_b8)
        if (c + 1 < NCHUNK) {
            const int n = tid * 3;
            __builtin_prefetch(W1t + (size_t)n * K4 + kc + 32, 0, 1);
        }

#if USE_ASYNC_LDS
        WAIT_ASYNC0();                  // our async copies landed in LDS
#endif
        __syncthreads();                // join all waves' slices

        // ---- WMMA: 2 M-tiles x 6 N-tiles per wave -------------------------
        Frag a0, a1;
        a0.q[0] = *(const uint4*)(Alds + lm        * LSTRIDE + kb);
        a0.q[1] = *(const uint4*)(Alds + lm        * LSTRIDE + kb + 16);
        a1.q[0] = *(const uint4*)(Alds + (16 + lm) * LSTRIDE + kb);
        a1.q[1] = *(const uint4*)(Alds + (16 + lm) * LSTRIDE + kb + 16);
        #pragma unroll
        for (int nt = 0; nt < NT_PER_WAVE; ++nt) {
            const int n = (wv * NT_PER_WAVE + nt) * 16 + lm;
            Frag bf;
            bf.q[0] = *(const uint4*)(Blds + n * LSTRIDE + kb);
            bf.q[1] = *(const uint4*)(Blds + n * LSTRIDE + kb + 16);
            acc[0][nt] = __builtin_amdgcn_wmma_f32_16x16x32_bf16(
                false, a0.v, false, bf.v, (short)0, acc[0][nt], false, false);
            acc[1][nt] = __builtin_amdgcn_wmma_f32_16x16x32_bf16(
                false, a1.v, false, bf.v, (short)0, acc[1][nt], false, false);
        }
        __syncthreads();
    }

    // ---- epilogue: bias + exact GELU + dot with W2 ------------------------
    float p[2][8];
    #pragma unroll
    for (int mt = 0; mt < 2; ++mt)
        #pragma unroll
        for (int v = 0; v < 8; ++v) p[mt][v] = 0.f;

    #pragma unroll
    for (int nt = 0; nt < NT_PER_WAVE; ++nt) {
        const int col = (wv * NT_PER_WAVE + nt) * 16 + lm;
        const float bias = b1[col];
        const float w2v  = W2[col];
        #pragma unroll
        for (int mt = 0; mt < 2; ++mt)
            #pragma unroll
            for (int v = 0; v < 8; ++v) {
                float x = acc[mt][nt][v] + bias;
                float g = 0.5f * x * (1.0f + erff(x * 0.70710678118654752f));
                p[mt][v] = fmaf(g, w2v, p[mt][v]);
            }
    }

    // partials -> LDS (overlays B buffer; protected by trailing loop sync)
    #pragma unroll
    for (int mt = 0; mt < 2; ++mt)
        #pragma unroll
        for (int v = 0; v < 8; ++v) {
            const int row = mt * 16 + (hi ? 8 + v : v);
            red[row * 128 + wv * 16 + lm] = p[mt][v];
        }
    __syncthreads();

    if (tid < BM) {
        const int r = r0 + tid;
        if (r < P) {
            float s = 0.f;
            const float* rr = red + tid * 128;
            for (int i = 0; i < 128; ++i) s += rr[i];   // fixed order: deterministic
            logits[r] = s + b2[0];
        }
    }
}

// ---------------------------------------------------------------------------
// Kernel 2/3: deterministic two-stage BCE-with-logits mean reduction
// ---------------------------------------------------------------------------
#define NB_LOSS 128

__global__ void bce_partial_kernel(const float* __restrict__ logits,
                                   const int*   __restrict__ labels,
                                   int P, float* __restrict__ partials) {
    __shared__ float s[256];
    const int tid = threadIdx.x;
    float a = 0.f;
    for (int i = blockIdx.x * 256 + tid; i < P; i += NB_LOSS * 256) {
        const float l = logits[i];
        const float y = (float)labels[i];
        a += fmaxf(l, 0.f) - l * y + log1pf(expf(-fabsf(l)));
    }
    s[tid] = a;
    __syncthreads();
    for (int off = 128; off > 0; off >>= 1) {
        if (tid < off) s[tid] += s[tid + off];
        __syncthreads();
    }
    if (tid == 0) partials[blockIdx.x] = s[0];
}

__global__ void bce_final_kernel(const float* __restrict__ partials,
                                 float* __restrict__ loss, int P) {
    __shared__ float s[NB_LOSS];
    const int tid = threadIdx.x;
    s[tid] = partials[tid];
    __syncthreads();
    for (int off = NB_LOSS / 2; off > 0; off >>= 1) {
        if (tid < off) s[tid] += s[tid + off];
        __syncthreads();
    }
    if (tid == 0) loss[0] = s[0] / (float)P;
}

// ---------------------------------------------------------------------------
extern "C" void kernel_launch(void* const* d_in, const int* in_sizes, int n_in,
                              void* d_out, int out_size, void* d_ws, size_t ws_size,
                              hipStream_t stream) {
    const float* emb    = (const float*)d_in[0];
    const int*   pa     = (const int*)d_in[1];
    const int*   pb     = (const int*)d_in[2];
    const int*   labels = (const int*)d_in[3];
    const float* W1     = (const float*)d_in[4];
    const float* b1     = (const float*)d_in[5];
    const float* W2     = (const float*)d_in[6];
    const float* b2     = (const float*)d_in[7];

    float* out    = (float*)d_out;
    float* loss   = out;          // d_out[0]
    float* logits = out + 1;      // d_out[1..P]

    const int P = in_sizes[1];

    // workspace: bf16 transposed W1 (768*3072*2 = 4,718,592 B) + 128 partials
    unsigned short* W1t = (unsigned short*)d_ws;
    float* partials = (float*)((char*)d_ws + (size_t)H * K4 * 2);

    convert_w1_kernel<<<(K4 * H) / 256, 256, 0, stream>>>(W1, W1t);

    const int nblk = (P + BM - 1) / BM;
    pair_head_gemm_kernel<<<nblk, 256, 0, stream>>>(
        emb, pa, pb, W1t, b1, W2, b2, logits, P);

    bce_partial_kernel<<<NB_LOSS, 256, 0, stream>>>(logits, labels, P, partials);
    bce_final_kernel<<<1, NB_LOSS, 0, stream>>>(partials, loss, P);
}
